// LADANForTextClassification_70368744177855
// MI455X (gfx1250) — compile-verified, matrix operand order checked by reading, not run on hardware
//
#include <hip/hip_runtime.h>
#include <hip/hip_bf16.h>
#include <math.h>

#define Bz 64
#define Sz 256
#define Hz 768
#define Cz 12
#define Lz 120
#define Ez 2400
#define HDz 384
#define G4z 1536   // 4*HD == 2*H

typedef __attribute__((ext_vector_type(8)))  _Float16 v8h;
typedef __attribute__((ext_vector_type(16))) _Float16 v16h;
typedef __attribute__((ext_vector_type(8)))  float    v8f;

// ---- WMMA fragment loader -------------------------------------------------
// Row-major f16 matrix, one 16x32 fragment (A) or K-major weight-row (B of W^T).
// lane L: half = L>>4, rr = L&15. 16-bit layout (ISA 7.12.2):
//   lanes 0-15 hold K = kb..kb+7 and kb+16..kb+23
//   lanes 16-31 hold K = kb+8..kb+15 and kb+24..kb+31
__device__ __forceinline__ v16h frag_row(const _Float16* rowp, int kb, int half) {
  const _Float16* p = rowp + kb + half * 8;
  v8h lo = *(const v8h*)(p);
  v8h hi = *(const v8h*)(p + 16);
  v16h r;
#pragma unroll
  for (int i = 0; i < 8; ++i) { r[i] = lo[i]; r[i + 8] = hi[i]; }
  return r;
}

__device__ __forceinline__ float sigf(float x) { return 1.0f / (1.0f + expf(-x)); }
__device__ __forceinline__ float geluf(float x) { return 0.5f * x * (1.0f + erff(x * 0.70710678118654752f)); }

// ---- utility kernels ------------------------------------------------------
__global__ void k_zero_f32(float* p, int n) {
  int i = blockIdx.x * blockDim.x + threadIdx.x;
  if (i < n) p[i] = 0.0f;
}
__global__ void k_zero_f16(_Float16* p, int n) {
  int i = blockIdx.x * blockDim.x + threadIdx.x;
  if (i < n) p[i] = (_Float16)0.0f;
}
__global__ void k_cvt(const float* __restrict__ s, _Float16* __restrict__ d, int n) {
  int i = blockIdx.x * blockDim.x + threadIdx.x;
  if (i < n) d[i] = (_Float16)s[i];
}
// label hidden states (120xH) -> padded 128xH f16, pad rows zero
__global__ void k_cvt_pad_ld(const float* __restrict__ s, _Float16* __restrict__ d) {
  int i = blockIdx.x * blockDim.x + threadIdx.x;
  if (i < 128 * Hz) {
    int r = i / Hz;
    d[i] = (r < Lz) ? (_Float16)s[i] : (_Float16)0.0f;
  }
}

// ---- graph degree ---------------------------------------------------------
__global__ void k_deg(const long long* __restrict__ eidx, float* __restrict__ deg_inv) {
  __shared__ int degs[Lz];
  int tid = threadIdx.x;
  for (int i = tid; i < Lz; i += blockDim.x) degs[i] = 0;
  __syncthreads();
  const long long* col = eidx + Ez;
  for (int e = tid; e < Ez; e += blockDim.x) atomicAdd(&degs[(int)col[e]], 1);
  __syncthreads();
  for (int i = tid; i < Lz; i += blockDim.x) {
    int d = degs[i];
    deg_inv[i] = (d > 0) ? 1.0f / (float)d : 0.0f;
  }
}

// ---- GDL edge message GEMM: agg[col[e]] += norm[e] * (cat(x[col],x[row]) @ lin2^T)
// tiles: 150 (edges/16) x 48 (H/16), one wave per tile, K = 2H = 1536
__global__ void k_gdl_msg(const _Float16* __restrict__ x, const _Float16* __restrict__ lin2,
                          const long long* __restrict__ eidx, const float* __restrict__ deg_inv,
                          float* __restrict__ agg) {
  int wave = blockIdx.x * (blockDim.x >> 5) + (threadIdx.x >> 5);
  int lane = threadIdx.x & 31, half = lane >> 4, rr = lane & 15;
  int mt = wave / (Hz / 16);
  int nt = wave % (Hz / 16);
  const long long* rowi = eidx;
  const long long* coli = eidx + Ez;
  int e = mt * 16 + rr;
  const _Float16* xc = x + (size_t)((int)coli[e]) * Hz;
  const _Float16* xr = x + (size_t)((int)rowi[e]) * Hz;
  const _Float16* wrow = lin2 + (size_t)(nt * 16 + rr) * (2 * Hz);
  v8f acc = {};
  for (int kb = 0; kb < 2 * Hz; kb += 32) {
    const _Float16* src = (kb < Hz) ? xc : xr;
    int k0 = (kb < Hz) ? kb : kb - Hz;
    v16h a = frag_row(src, k0, half);
    v16h b = frag_row(wrow, kb, half);
    acc = __builtin_amdgcn_wmma_f32_16x16x32_f16(false, a, false, b, (short)0, acc, false, false);
  }
#pragma unroll
  for (int r = 0; r < 8; ++r) {
    int e2 = mt * 16 + r + half * 8;
    int ce = (int)coli[e2];
    float nv = deg_inv[(int)rowi[e2]];
    atomicAdd(&agg[(size_t)ce * Hz + nt * 16 + rr], acc[r] * nv);
  }
}

// ---- GDL combine: out = gelu(x @ lin1^T - agg + bias). M padded to 128.
__global__ void k_gdl_combine(const _Float16* __restrict__ x, const _Float16* __restrict__ lin1,
                              const float* __restrict__ agg, const float* __restrict__ bias,
                              _Float16* __restrict__ of16, float* __restrict__ of32) {
  int wave = blockIdx.x * (blockDim.x >> 5) + (threadIdx.x >> 5);
  int lane = threadIdx.x & 31, half = lane >> 4, rr = lane & 15;
  int mt = wave / (Hz / 16);       // 0..7
  int nt = wave % (Hz / 16);       // 0..47
  int row = mt * 16 + rr;
  const _Float16* arow = x + (size_t)row * Hz;
  const _Float16* brow = lin1 + (size_t)(nt * 16 + rr) * Hz;
  v8f acc = {};
  for (int kb = 0; kb < Hz; kb += 32) {
    v16h a = frag_row(arow, kb, half);
    v16h b = frag_row(brow, kb, half);
    acc = __builtin_amdgcn_wmma_f32_16x16x32_f16(false, a, false, b, (short)0, acc, false, false);
  }
#pragma unroll
  for (int r = 0; r < 8; ++r) {
    int m = mt * 16 + r + half * 8;
    if (m < Lz) {
      int n = nt * 16 + rr;
      float v = acc[r] - agg[(size_t)m * Hz + n] + bias[n];
      v = geluf(v);
      if (of16) of16[(size_t)m * Hz + n] = (_Float16)v;
      if (of32) of32[(size_t)m * Hz + n] = v;
    }
  }
}

// ---- community masked max/min pooling ------------------------------------
__global__ void k_minmax(const float* __restrict__ ld2, const unsigned char* __restrict__ c2l,
                         float* __restrict__ mmax, float* __restrict__ mmin) {
  int c = blockIdx.x;
  for (int h = threadIdx.x; h < Hz; h += blockDim.x) {
    float mx = -INFINITY, mn = INFINITY;
    for (int l = 0; l < Lz; ++l)
      if (c2l[c * Lz + l]) {
        float v = ld2[(size_t)l * Hz + h];
        mx = fmaxf(mx, v);
        mn = fminf(mn, v);
      }
    mmax[(size_t)c * Hz + h] = mx;
    mmin[(size_t)c * Hz + h] = mn;
  }
}

// ctx = cat(mmax,mmin) @ ctx_W^T + ctx_b  (12 x 768), stored f16 padded to 16 rows
__global__ void k_ctx(const float* __restrict__ mmax, const float* __restrict__ mmin,
                      const float* __restrict__ W, const float* __restrict__ bb,
                      _Float16* __restrict__ ctx16) {
  int c = blockIdx.x;
  for (int o = threadIdx.x; o < Hz; o += blockDim.x) {
    const float* wr = W + (size_t)o * (2 * Hz);
    float s = bb[o];
    for (int k = 0; k < Hz; ++k) s += mmax[(size_t)c * Hz + k] * wr[k];
    for (int k = 0; k < Hz; ++k) s += mmin[(size_t)c * Hz + k] * wr[Hz + k];
    ctx16[(size_t)c * Hz + o] = (_Float16)s;
  }
}

// ---- LSTM input projection: pre[dir][s][b][n] = x[b][s] @ Wih^T + b ------
// 4x M-blocked: one wave computes 4 M-tiles against one shared B fragment.
__global__ void k_pre_gemm(const _Float16* __restrict__ x16, const _Float16* __restrict__ Wih16,
                           const float* __restrict__ bf, const float* __restrict__ br,
                           float* __restrict__ pre) {
  int wave = blockIdx.x * (blockDim.x >> 5) + (threadIdx.x >> 5);
  int lane = threadIdx.x & 31, half = lane >> 4, rr = lane & 15;
  const int MB = (Bz * Sz) / 64, NT = G4z / 16;   // 256 m-blocks of 4 tiles, 96 n-tiles
  int dir = wave / (MB * NT);
  int rem = wave % (MB * NT);
  int mb = rem / NT, nt = rem % NT;
  const _Float16* arow[4];
#pragma unroll
  for (int i = 0; i < 4; ++i)
    arow[i] = x16 + (size_t)((mb * 4 + i) * 16 + rr) * Hz;
  const _Float16* brow = Wih16 + (size_t)dir * G4z * Hz + (size_t)(nt * 16 + rr) * Hz;
  v8f acc[4] = {};
  for (int kb = 0; kb < Hz; kb += 32) {
    v16h b = frag_row(brow, kb, half);
#pragma unroll
    for (int i = 0; i < 4; ++i) {
      v16h a = frag_row(arow[i], kb, half);
      acc[i] = __builtin_amdgcn_wmma_f32_16x16x32_f16(false, a, false, b, (short)0, acc[i], false, false);
    }
  }
  const float* bias = dir ? br : bf;
#pragma unroll
  for (int i = 0; i < 4; ++i) {
#pragma unroll
    for (int r = 0; r < 8; ++r) {
      int m = (mb * 4 + i) * 16 + r + half * 8;
      int b_ = m / Sz, s_ = m % Sz;
      int n = nt * 16 + rr;
      pre[(((size_t)dir * Sz + s_) * Bz + b_) * G4z + n] = acc[i][r] + bias[n];
    }
  }
}

// ---- LSTM per-step gate GEMM: g = pre[t] + h_prev @ Whh^T ----------------
// one tile per wave: keeps 768 waves of parallelism in the latency-critical scan
__global__ void k_lstm_gemm(const _Float16* __restrict__ h16, const _Float16* __restrict__ Whh16,
                            const float* __restrict__ pre, float* __restrict__ g, int t) {
  int wave = blockIdx.x * (blockDim.x >> 5) + (threadIdx.x >> 5);
  int lane = threadIdx.x & 31, half = lane >> 4, rr = lane & 15;
  const int MT = Bz / 16, NT = G4z / 16;   // 4, 96
  int dir = wave / (MT * NT);
  int rem = wave % (MT * NT);
  int nt = rem / MT, mt = rem % MT;
  int s_t = dir ? (Sz - 1 - t) : t;
  const _Float16* arow = h16 + (size_t)dir * Bz * HDz + (size_t)(mt * 16 + rr) * HDz;
  const _Float16* brow = Whh16 + (size_t)dir * G4z * HDz + (size_t)(nt * 16 + rr) * HDz;
  v8f acc = {};
  for (int kb = 0; kb < HDz; kb += 32) {
    v16h a = frag_row(arow, kb, half);
    v16h b = frag_row(brow, kb, half);
    acc = __builtin_amdgcn_wmma_f32_16x16x32_f16(false, a, false, b, (short)0, acc, false, false);
  }
#pragma unroll
  for (int r = 0; r < 8; ++r) {
    int m = mt * 16 + r + half * 8;
    int n = nt * 16 + rr;
    g[((size_t)dir * Bz + m) * G4z + n] =
        acc[r] + pre[(((size_t)dir * Sz + s_t) * Bz + m) * G4z + n];
  }
}

// ---- LSTM pointwise gate update ------------------------------------------
__global__ void k_lstm_point(const float* __restrict__ g, float* __restrict__ cbuf,
                             _Float16* __restrict__ h16, float* __restrict__ out32,
                             _Float16* __restrict__ out16, int t) {
  int idx = blockIdx.x * blockDim.x + threadIdx.x;
  if (idx >= 2 * Bz * HDz) return;
  int dir = idx / (Bz * HDz);
  int rem = idx % (Bz * HDz);
  int b_ = rem / HDz, j = rem % HDz;
  const float* gb = g + ((size_t)dir * Bz + b_) * G4z;
  float gi = gb[j], gf = gb[HDz + j], gg = gb[2 * HDz + j], go = gb[3 * HDz + j];
  float c = cbuf[idx];
  c = sigf(gf) * c + sigf(gi) * tanhf(gg);
  float h = sigf(go) * tanhf(c);
  cbuf[idx] = c;
  h16[idx] = (_Float16)h;
  int s_t = dir ? (Sz - 1 - t) : t;
  size_t o = ((size_t)b_ * Sz + s_t) * Hz + dir * HDz + j;
  out32[o] = h;
  out16[o] = (_Float16)h;
}

// ---- attention projection: activated = tanh(outputs @ attn_W^T + b) -> f16
// 4x M-blocked like k_pre_gemm.
__global__ void k_attn_gemm(const _Float16* __restrict__ o16, const _Float16* __restrict__ W16,
                            const float* __restrict__ bb, _Float16* __restrict__ act16) {
  int wave = blockIdx.x * (blockDim.x >> 5) + (threadIdx.x >> 5);
  int lane = threadIdx.x & 31, half = lane >> 4, rr = lane & 15;
  const int NT = Hz / 16;   // 48
  int mb = wave / NT, nt = wave % NT;   // mb: 0..255
  const _Float16* arow[4];
#pragma unroll
  for (int i = 0; i < 4; ++i)
    arow[i] = o16 + (size_t)((mb * 4 + i) * 16 + rr) * Hz;
  const _Float16* brow = W16 + (size_t)(nt * 16 + rr) * Hz;
  v8f acc[4] = {};
  for (int kb = 0; kb < Hz; kb += 32) {
    v16h b = frag_row(brow, kb, half);
#pragma unroll
    for (int i = 0; i < 4; ++i) {
      v16h a = frag_row(arow[i], kb, half);
      acc[i] = __builtin_amdgcn_wmma_f32_16x16x32_f16(false, a, false, b, (short)0, acc[i], false, false);
    }
  }
#pragma unroll
  for (int i = 0; i < 4; ++i) {
#pragma unroll
    for (int r = 0; r < 8; ++r) {
      int m = (mb * 4 + i) * 16 + r + half * 8;
      int n = nt * 16 + rr;
      act16[(size_t)m * Hz + n] = (_Float16)tanhf(acc[i][r] + bb[n]);
    }
  }
}

// ---- scores: activated(BS x H) @ ctx^T(H x 16pad) -> scores[c][b][s] -----
__global__ void k_score_gemm(const _Float16* __restrict__ act16, const _Float16* __restrict__ ctx16,
                             float* __restrict__ scores) {
  int wave = blockIdx.x * (blockDim.x >> 5) + (threadIdx.x >> 5);   // = mtile
  int lane = threadIdx.x & 31, half = lane >> 4, rr = lane & 15;
  const _Float16* arow = act16 + (size_t)(wave * 16 + rr) * Hz;
  const _Float16* brow = ctx16 + (size_t)rr * Hz;
  v8f acc = {};
  for (int kb = 0; kb < Hz; kb += 32) {
    v16h a = frag_row(arow, kb, half);
    v16h b = frag_row(brow, kb, half);
    acc = __builtin_amdgcn_wmma_f32_16x16x32_f16(false, a, false, b, (short)0, acc, false, false);
  }
  if (rr < Cz) {
#pragma unroll
    for (int r = 0; r < 8; ++r) {
      int m = wave * 16 + r + half * 8;
      int b_ = m / Sz, s_ = m % Sz;
      scores[((size_t)rr * Bz + b_) * Sz + s_] = acc[r];
    }
  }
}

// ---- softmax over s, per (c,b); reference masks with -1e-32 ---------------
__global__ void k_softmax(float* __restrict__ sc, const unsigned char* __restrict__ mask) {
  __shared__ float red[Sz];
  int cb = blockIdx.x;
  int b_ = cb % Bz;
  int s_ = threadIdx.x;
  float v = sc[(size_t)cb * Sz + s_];
  v = mask[b_ * Sz + s_] ? v : -1e-32f;
  red[s_] = v;
  __syncthreads();
  for (int st = Sz / 2; st > 0; st >>= 1) {
    if (s_ < st) red[s_] = fmaxf(red[s_], red[s_ + st]);
    __syncthreads();
  }
  float mx = red[0];
  __syncthreads();
  float e = expf(v - mx);
  red[s_] = e;
  __syncthreads();
  for (int st = Sz / 2; st > 0; st >>= 1) {
    if (s_ < st) red[s_] += red[s_ + st];
    __syncthreads();
  }
  sc[(size_t)cb * Sz + s_] = e / red[0];
}

// ---- hid = sum_s w[c,b,s]*out[b,s,:]; final = basic + sum_c cl[b,c]*hid ---
__global__ void k_hid_final(const float* __restrict__ w, const float* __restrict__ out32,
                            const float* __restrict__ cl, const float* __restrict__ basic,
                            float* __restrict__ fin) {
  int b_ = blockIdx.x;
  int tid = threadIdx.x;   // 256 threads, 3 h each
  float acc[Cz][3];
#pragma unroll
  for (int c = 0; c < Cz; ++c)
#pragma unroll
    for (int u = 0; u < 3; ++u) acc[c][u] = 0.0f;
  for (int s_ = 0; s_ < Sz; ++s_) {
    float ov[3];
#pragma unroll
    for (int u = 0; u < 3; ++u) ov[u] = out32[((size_t)b_ * Sz + s_) * Hz + tid + u * 256];
#pragma unroll
    for (int c = 0; c < Cz; ++c) {
      float wv = w[((size_t)c * Bz + b_) * Sz + s_];
#pragma unroll
      for (int u = 0; u < 3; ++u) acc[c][u] += wv * ov[u];
    }
  }
#pragma unroll
  for (int u = 0; u < 3; ++u) {
    int h = tid + u * 256;
    float f = basic[(size_t)b_ * Hz + h];
#pragma unroll
    for (int c = 0; c < Cz; ++c) f += cl[b_ * Cz + c] * acc[c][u];
    fin[(size_t)b_ * Hz + h] = f;
  }
}

// ---- community logits + BCE sum (single block of 768 = 64*12) -------------
__global__ void k_comm(const float* __restrict__ basic, const float* __restrict__ W,
                       const float* __restrict__ bb, const float* __restrict__ y,
                       float* __restrict__ out, float* __restrict__ acc) {
  __shared__ float red[Bz * Cz];
  int tid = threadIdx.x;
  int b_ = tid / Cz, c = tid % Cz;
  const float* br = basic + (size_t)b_ * Hz;
  const float* wr = W + (size_t)c * Hz;
  float z = bb[c];
  for (int k = 0; k < Hz; ++k) z += br[k] * wr[k];
  out[tid] = z;
  float yy = y[tid];
  red[tid] = fmaxf(z, 0.0f) - z * yy + log1pf(expf(-fabsf(z)));
  __syncthreads();
  for (int st = 512; st > 0; st >>= 1) {
    if (tid < st && tid + st < Bz * Cz) red[tid] += red[tid + st];
    __syncthreads();
  }
  if (tid == 0) acc[0] = red[0];
}

// ---- label logits + BCE sum (7680 threads) --------------------------------
__global__ void k_cls(const float* __restrict__ fin, const float* __restrict__ W,
                      const float* __restrict__ bb, const float* __restrict__ y,
                      float* __restrict__ out, float* __restrict__ acc) {
  __shared__ float red[256];
  int idx = blockIdx.x * blockDim.x + threadIdx.x;
  int b_ = idx / Lz, l = idx % Lz;
  const float* fr = fin + (size_t)b_ * Hz;
  const float* wr = W + (size_t)l * Hz;
  float z = bb[l];
  for (int k = 0; k < Hz; ++k) z += fr[k] * wr[k];
  out[idx] = z;
  float yy = y[idx];
  red[threadIdx.x] = fmaxf(z, 0.0f) - z * yy + log1pf(expf(-fabsf(z)));
  __syncthreads();
  for (int st = 128; st > 0; st >>= 1) {
    if (threadIdx.x < st) red[threadIdx.x] += red[threadIdx.x + st];
    __syncthreads();
  }
  if (threadIdx.x == 0) atomicAdd(&acc[1], red[0]);
}

__global__ void k_loss(const float* __restrict__ acc, float* __restrict__ out0) {
  if (threadIdx.x == 0)
    out0[0] = 0.5f * (acc[0] / (float)(Bz * Cz)) + acc[1] / (float)(Bz * Lz);
}

// ===========================================================================
extern "C" void kernel_launch(void* const* d_in, const int* in_sizes, int n_in,
                              void* d_out, int out_size, void* d_ws, size_t ws_size,
                              hipStream_t stream) {
  const float* x_in   = (const float*)d_in[0];
  const float* basic  = (const float*)d_in[1];
  const float* ldh    = (const float*)d_in[2];
  const float* clab   = (const float*)d_in[3];
  const float* labels = (const float*)d_in[4];
  const unsigned char* smask = (const unsigned char*)d_in[5];
  const long long* eidx = (const long long*)d_in[6];
  const unsigned char* c2l = (const unsigned char*)d_in[7];
  const float* g1l1 = (const float*)d_in[8];
  const float* g1l2 = (const float*)d_in[9];
  const float* g1b  = (const float*)d_in[10];
  const float* g2l1 = (const float*)d_in[11];
  const float* g2l2 = (const float*)d_in[12];
  const float* g2b  = (const float*)d_in[13];
  const float* ctxW = (const float*)d_in[14];
  const float* ctxb = (const float*)d_in[15];
  const float* attnW = (const float*)d_in[16];
  const float* attnb = (const float*)d_in[17];
  const float* WihF = (const float*)d_in[18];
  const float* WhhF = (const float*)d_in[19];
  const float* bF   = (const float*)d_in[20];
  const float* WihR = (const float*)d_in[21];
  const float* WhhR = (const float*)d_in[22];
  const float* bR   = (const float*)d_in[23];
  const float* commW = (const float*)d_in[24];
  const float* commb = (const float*)d_in[25];
  const float* clsW = (const float*)d_in[26];
  const float* clsb = (const float*)d_in[27];
  float* out = (float*)d_out;   // [0]=loss, [1..7681)=logits, then comm_logits

  // ---- workspace layout (~343 MB) ----
  size_t off = 0;
  auto alloc = [&](size_t bytes) -> void* {
    void* p = (void*)((char*)d_ws + off);
    off += (bytes + 255) & ~(size_t)255;
    return p;
  };
  float*     deg_inv = (float*)alloc(128 * 4);
  float*     agg     = (float*)alloc((size_t)Lz * Hz * 4);
  _Float16*  ldA     = (_Float16*)alloc((size_t)128 * Hz * 2);
  _Float16*  ldB     = (_Float16*)alloc((size_t)128 * Hz * 2);
  float*     ld2     = (float*)alloc((size_t)Lz * Hz * 4);
  float*     mmax    = (float*)alloc((size_t)Cz * Hz * 4);
  float*     mmin    = (float*)alloc((size_t)Cz * Hz * 4);
  _Float16*  ctx16   = (_Float16*)alloc((size_t)16 * Hz * 2);
  _Float16*  x16     = (_Float16*)alloc((size_t)Bz * Sz * Hz * 2);
  _Float16*  l2a     = (_Float16*)alloc((size_t)Hz * 2 * Hz * 2);
  _Float16*  l2b     = (_Float16*)alloc((size_t)Hz * 2 * Hz * 2);
  _Float16*  l1a     = (_Float16*)alloc((size_t)Hz * Hz * 2);
  _Float16*  l1b     = (_Float16*)alloc((size_t)Hz * Hz * 2);
  _Float16*  Wih16   = (_Float16*)alloc((size_t)2 * G4z * Hz * 2);
  _Float16*  Whh16   = (_Float16*)alloc((size_t)2 * G4z * HDz * 2);
  _Float16*  aW16    = (_Float16*)alloc((size_t)Hz * Hz * 2);
  float*     pre     = (float*)alloc((size_t)2 * Sz * Bz * G4z * 4);
  float*     gbuf    = (float*)alloc((size_t)2 * Bz * G4z * 4);
  float*     cbuf    = (float*)alloc((size_t)2 * Bz * HDz * 4);
  _Float16*  hbuf    = (_Float16*)alloc((size_t)2 * Bz * HDz * 2);
  float*     out32   = (float*)alloc((size_t)Bz * Sz * Hz * 4);
  _Float16*  out16   = (_Float16*)alloc((size_t)Bz * Sz * Hz * 2);
  _Float16*  act16   = (_Float16*)alloc((size_t)Bz * Sz * Hz * 2);
  float*     scores  = (float*)alloc((size_t)Cz * Bz * Sz * 4);
  float*     fin     = (float*)alloc((size_t)Bz * Hz * 4);
  float*     accv    = (float*)alloc(256);
  (void)ws_size; (void)in_sizes; (void)n_in; (void)out_size;

  auto gs = [](int n) { return dim3((n + 255) / 256); };

  // ---- deterministic zero-init of state/accumulators ----
  k_zero_f32<<<gs(Lz * Hz), 256, 0, stream>>>(agg, Lz * Hz);
  k_zero_f32<<<gs(2 * Bz * HDz), 256, 0, stream>>>(cbuf, 2 * Bz * HDz);
  k_zero_f32<<<gs(64), 256, 0, stream>>>(accv, 64);
  k_zero_f16<<<gs(128 * Hz), 256, 0, stream>>>(ldB, 128 * Hz);
  k_zero_f16<<<gs(2 * Bz * HDz), 256, 0, stream>>>(hbuf, 2 * Bz * HDz);
  k_zero_f16<<<gs(16 * Hz), 256, 0, stream>>>(ctx16, 16 * Hz);

  // ---- f32 -> f16 conversions ----
  k_cvt<<<gs(Bz * Sz * Hz), 256, 0, stream>>>(x_in, x16, Bz * Sz * Hz);
  k_cvt<<<gs(Hz * 2 * Hz), 256, 0, stream>>>(g1l2, l2a, Hz * 2 * Hz);
  k_cvt<<<gs(Hz * 2 * Hz), 256, 0, stream>>>(g2l2, l2b, Hz * 2 * Hz);
  k_cvt<<<gs(Hz * Hz), 256, 0, stream>>>(g1l1, l1a, Hz * Hz);
  k_cvt<<<gs(Hz * Hz), 256, 0, stream>>>(g2l1, l1b, Hz * Hz);
  k_cvt<<<gs(G4z * Hz), 256, 0, stream>>>(WihF, Wih16, G4z * Hz);
  k_cvt<<<gs(G4z * Hz), 256, 0, stream>>>(WihR, Wih16 + (size_t)G4z * Hz, G4z * Hz);
  k_cvt<<<gs(G4z * HDz), 256, 0, stream>>>(WhhF, Whh16, G4z * HDz);
  k_cvt<<<gs(G4z * HDz), 256, 0, stream>>>(WhhR, Whh16 + (size_t)G4z * HDz, G4z * HDz);
  k_cvt<<<gs(Hz * Hz), 256, 0, stream>>>(attnW, aW16, Hz * Hz);
  k_cvt_pad_ld<<<gs(128 * Hz), 256, 0, stream>>>(ldh, ldA);

  // ---- graph degree + GDL layers ----
  k_deg<<<1, 256, 0, stream>>>(eidx, deg_inv);
  k_gdl_msg<<<(150 * 48) / 4, 128, 0, stream>>>(ldA, l2a, eidx, deg_inv, agg);
  k_gdl_combine<<<(8 * 48) / 4, 128, 0, stream>>>(ldA, l1a, agg, g1b, ldB, (float*)nullptr);
  k_zero_f32<<<gs(Lz * Hz), 256, 0, stream>>>(agg, Lz * Hz);
  k_gdl_msg<<<(150 * 48) / 4, 128, 0, stream>>>(ldB, l2b, eidx, deg_inv, agg);
  k_gdl_combine<<<(8 * 48) / 4, 128, 0, stream>>>(ldB, l1b, agg, g2b, (_Float16*)nullptr, ld2);

  // ---- community context ----
  k_minmax<<<Cz, 256, 0, stream>>>(ld2, c2l, mmax, mmin);
  k_ctx<<<Cz, 256, 0, stream>>>(mmax, mmin, ctxW, ctxb, ctx16);

  // ---- LSTM: input projection then 256-step scan (both directions fused) ----
  k_pre_gemm<<<(256 * 96 * 2) / 4, 128, 0, stream>>>(x16, Wih16, bF, bR, pre);
  for (int t = 0; t < Sz; ++t) {
    k_lstm_gemm<<<(2 * 4 * 96) / 4, 128, 0, stream>>>(hbuf, Whh16, pre, gbuf, t);
    k_lstm_point<<<(2 * Bz * HDz) / 256, 256, 0, stream>>>(gbuf, cbuf, hbuf, out32, out16, t);
  }

  // ---- attention + pooling + heads + losses ----
  k_attn_gemm<<<(256 * 48) / 4, 128, 0, stream>>>(out16, aW16, attnb, act16);
  k_score_gemm<<<1024 / 4, 128, 0, stream>>>(act16, ctx16, scores);
  k_softmax<<<Cz * Bz, Sz, 0, stream>>>(scores, smask);
  k_hid_final<<<Bz, 256, 0, stream>>>(scores, out32, clab, basic, fin);
  k_comm<<<1, Bz * Cz, 0, stream>>>(basic, commW, commb, clab, out + 1 + Bz * Lz, accv);
  k_cls<<<(Bz * Lz) / 256, 256, 0, stream>>>(fin, clsW, clsb, labels, out + 1, accv);
  k_loss<<<1, 1, 0, stream>>>(accv, out);
}
